// FullRankLeakyNoisyRNN_91036126806940
// MI455X (gfx1250) — matrix operand order repaired and checked by reading, not compiled, not run on hardware
//
#include <hip/hip_runtime.h>
#include <cstdint>

// Problem constants (match reference)
#define B_TOT 128
#define T_LEN 1000
#define I_SZ  4
#define H_SZ  512
#define O_SZ  2
#define ALPHA_C 0.2f
#define NSTD_C  0.05f

typedef __attribute__((ext_vector_type(16))) __bf16 v16bf;
typedef __attribute__((ext_vector_type(8)))  __bf16 v8bf;
typedef __attribute__((ext_vector_type(4)))  __bf16 v4bf;
typedef __attribute__((ext_vector_type(8)))  float  v8f;
typedef __attribute__((ext_vector_type(4)))  float  v4f;

#define RSTRIDE 520   // bf16 elems per row of r tile (1040 B rows keep A-frag loads conflict-free)
#define HSTRIDE 512   // f32 elems per row of h tile
#define NSTR    516   // f32 elems per row of noise stage (+4 pad kills lane/lane+16 collisions)

// Fragment-swizzled w_rec layout:
//   B fragment (nt, kc): cols [nt*16, nt*16+16), K [kc*32, kc*32+32)
//   1 KB per fragment stored as [half(2)][lane(32)][8 x bf16]
//   element (lane l, j): col = nt*16 + (l&15), k = kc*32 + (l>>4)*16 + j
//   kc<7 fragments packed first (224 KB, LDS-resident); kc>=7 after (L2-streamed).
#define NFRAG_ELEMS 512
#define NT_COUNT    32
#define LOWK        7
#define HIK         9
#define LOW_FRAGS   (NT_COUNT * LOWK)                  // 224
#define LOW_ELEMS   (LOW_FRAGS * NFRAG_ELEMS)          // 114688 bf16 = 224 KB

__global__ __launch_bounds__(256) void wrec_swizzle(const float* __restrict__ w,
                                                    __bf16* __restrict__ o) {
    int i = blockIdx.x * blockDim.x + threadIdx.x;
    if (i >= H_SZ * H_SZ) return;
    const int inf  = i & 511;          // index within fragment
    const int frag = i >> 9;           // 0..511
    const int half = inf >> 8;
    const int l    = (inf >> 3) & 31;
    const int jj   = inf & 7;
    const int j    = half * 8 + jj;
    int nt, kc;
    if (frag < LOW_FRAGS) { nt = frag / LOWK; kc = frag % LOWK; }
    else { const int fr = frag - LOW_FRAGS; nt = fr / HIK; kc = LOWK + fr % HIK; }
    const int col = nt * 16 + (l & 15);
    const int k   = kc * 32 + (l >> 4) * 16 + j;
    o[i] = (__bf16)w[col * H_SZ + k];
}

__device__ __forceinline__ float fast_tanh(float x) {
    x = fminf(15.f, fmaxf(-15.f, x));
    const float e = __expf(2.f * x);
    return fmaf(-2.f, __builtin_amdgcn_rcpf(e + 1.f), 1.f);
}

// Persistent RNN: 8 blocks x 512 threads; each block owns 16 batch rows.
__global__ __launch_bounds__(512) void rnn_persistent(
    const float* __restrict__ input,     // (B,T,I)
    const float* __restrict__ noise,     // (B,T,H)
    const float* __restrict__ w_i,       // (I,H)
    const float* __restrict__ s_i,       // (I)
    const float* __restrict__ w_o,       // (H,O)
    const float* __restrict__ s_o,       // (O)
    const float* __restrict__ h0,        // (H)
    const __bf16* __restrict__ wrec_swz, // fragment-swizzled bf16 w_rec (512 KB)
    float* __restrict__ out)             // (B,T,O)
{
    __shared__ __align__(16) __bf16 wrec_l[LOW_ELEMS];        // 224 KB low-K B fragments
    __shared__ __align__(16) float  h_lds[16 * HSTRIDE];      //  32 KB state
    __shared__ __align__(16) __bf16 r_lds[16 * RSTRIDE];      //  16.25 KB tanh(h)
    __shared__ __align__(16) float  noise_stage[16 * NSTR];   //  32.25 KB async-DMA'd noise[t]
    __shared__ __align__(16) float  w_im[I_SZ * H_SZ];        //   8 KB  w_i * s_i
    __shared__ __align__(16) float  wo_l[O_SZ * H_SZ];        //   4 KB  [o][col] = w_o*s_o/H
    __shared__ __align__(16) float  in_stage[16 * I_SZ];      //   256 B

    const int tid  = (int)threadIdx.x;
    const int lane = tid & 31;
    const int wid  = tid >> 5;            // wave id 0..15
    const int b0   = (int)blockIdx.x * 16;

    // ---- one-time init: small tables + h0 broadcast ----
    {
        const int c = tid;                // 0..511
        #pragma unroll
        for (int i = 0; i < I_SZ; ++i)
            w_im[i * H_SZ + c] = w_i[i * H_SZ + c] * s_i[i];
        #pragma unroll
        for (int o = 0; o < O_SZ; ++o)
            wo_l[o * H_SZ + c] = w_o[c * O_SZ + o] * s_o[o] * (1.0f / (float)H_SZ);
        const float hv = h0[c];
        #pragma unroll
        for (int r = 0; r < 16; ++r)
            h_lds[r * HSTRIDE + c] = hv;
    }
    // ---- one-time init: linear 224 KB preload of low-K fragments ----
    {
        const v4f* src = (const v4f*)wrec_swz;
        v4f* dst = (v4f*)wrec_l;
        #pragma unroll 4
        for (int i = tid; i < (int)(LOW_ELEMS * sizeof(__bf16) / 16); i += 512)
            dst[i] = src[i];
    }
    __syncthreads();

    const int m   = lane & 15;
    const int hi  = lane >> 4;
    const int c0  = wid * 32;
    const int c1  = c0 + 16;
    const int nt0 = wid * 2;
    const int nt1 = nt0 + 1;
    const __bf16* ghi = wrec_swz + LOW_ELEMS;   // high-K fragments (L2-resident)

    // A fragment (16x32 bf16) from r_lds
    auto load_a = [&](int kchunk) -> v16bf {
        const int ka = kchunk * 32 + hi * 16;
        const v8bf alo = *(const v8bf*)&r_lds[m * RSTRIDE + ka];
        const v8bf ahi = *(const v8bf*)&r_lds[m * RSTRIDE + ka + 8];
        v16bf a;
        #pragma unroll
        for (int j = 0; j < 8; ++j) { a[j] = alo[j]; a[8 + j] = ahi[j]; }
        return a;
    };
    // B fragment from LDS cache (base = element index of lane's low half)
    auto load_b_lds = [&](int base) -> v16bf {
        const v8bf bl = *(const v8bf*)&wrec_l[base];
        const v8bf bh = *(const v8bf*)&wrec_l[base + 256];
        v16bf b;
        #pragma unroll
        for (int j = 0; j < 8; ++j) { b[j] = bl[j]; b[8 + j] = bh[j]; }
        return b;
    };
    // B fragment streamed from L2
    auto load_b_g = [&](const __bf16* p) -> v16bf {
        const v8bf bl = *(const v8bf*)p;
        const v8bf bh = *(const v8bf*)(p + 256);
        v16bf b;
        #pragma unroll
        for (int j = 0; j < 8; ++j) { b[j] = bl[j]; b[8 + j] = bh[j]; }
        return b;
    };

    for (int t = 0; t <= T_LEN; ++t) {
        // -------- Phase 1: async noise prefetch; r = tanh(h); out[t-1]; stage input --------
        {
            const int row = wid;
            if (t < T_LEN) {
                // async DMA this wave's noise region (rows 0..15, cols [c0,c0+32)) into LDS
                #pragma unroll
                for (int q = 0; q < 4; ++q) {
                    const int c    = lane + 32 * q;       // 0..127
                    const int nrow = c >> 3;
                    const int ncol = c0 + (c & 7) * 4;
                    const float* g = noise + ((size_t)(b0 + nrow) * T_LEN + (size_t)t) * H_SZ + ncol;
                    const uint32_t lo = (uint32_t)(uintptr_t)&noise_stage[nrow * NSTR + ncol];
                    asm volatile("global_load_async_to_lds_b128 %0, %1, off"
                                 :: "v"(lo), "v"(g) : "memory");
                }
            }
            float o0 = 0.f, o1 = 0.f;
            #pragma unroll
            for (int jb = 0; jb < 4; ++jb) {
                const int cb = lane * 4 + 128 * jb;       // conflict-free: lane stride 4 dwords
                const v4f hv = *(const v4f*)&h_lds[row * HSTRIDE + cb];
                float rv[4];
                #pragma unroll
                for (int i2 = 0; i2 < 4; ++i2) {
                    const float r = fast_tanh(hv[i2]);
                    rv[i2] = r;
                    o0 += r * wo_l[cb + i2];
                    o1 += r * wo_l[H_SZ + cb + i2];
                }
                if (t < T_LEN) {
                    v4bf p;
                    #pragma unroll
                    for (int i2 = 0; i2 < 4; ++i2) p[i2] = (__bf16)rv[i2];
                    *(v4bf*)&r_lds[row * RSTRIDE + cb] = p;  // b64, all 32 lanes distinct banks
                }
            }
            #pragma unroll
            for (int off = 16; off > 0; off >>= 1) {
                o0 += __shfl_xor(o0, off, 32);
                o1 += __shfl_xor(o1, off, 32);
            }
            if (t > 0 && lane == 0) {
                const size_t ob = ((size_t)(b0 + row) * T_LEN + (size_t)(t - 1)) * O_SZ;
                out[ob + 0] = o0;
                out[ob + 1] = o1;
            }
            if (t < T_LEN && tid < 16) {
                const float* ip = input + ((size_t)(b0 + tid) * T_LEN + (size_t)t) * I_SZ;
                *(v4f*)&in_stage[tid * I_SZ] = *(const v4f*)ip;
            }
        }
        if (t == T_LEN) break;
        __syncthreads();

        // -------- Phase 2: acc = r @ w_rec^T; 7 K-chunks from LDS, 9 from L2 --------
        v8f acc0 = {};
        v8f acc1 = {};
        for (int kc = 0; kc < LOWK; ++kc) {
            {   // low-K (LDS)
                const v16bf a = load_a(kc);
                const v16bf bv0 = load_b_lds((nt0 * LOWK + kc) * NFRAG_ELEMS + lane * 8);
                const v16bf bv1 = load_b_lds((nt1 * LOWK + kc) * NFRAG_ELEMS + lane * 8);
                acc0 = __builtin_amdgcn_wmma_f32_16x16x32_bf16(false, a, false, bv0,
                                                               (short)0, acc0, false, false);
                acc1 = __builtin_amdgcn_wmma_f32_16x16x32_bf16(false, a, false, bv1,
                                                               (short)0, acc1, false, false);
            }
            {   // high-K (L2), kh = kc + 7
                const v16bf a = load_a(kc + LOWK);
                const v16bf bv0 = load_b_g(ghi + (size_t)(nt0 * HIK + kc) * NFRAG_ELEMS + lane * 8);
                const v16bf bv1 = load_b_g(ghi + (size_t)(nt1 * HIK + kc) * NFRAG_ELEMS + lane * 8);
                acc0 = __builtin_amdgcn_wmma_f32_16x16x32_bf16(false, a, false, bv0,
                                                               (short)0, acc0, false, false);
                acc1 = __builtin_amdgcn_wmma_f32_16x16x32_bf16(false, a, false, bv1,
                                                               (short)0, acc1, false, false);
            }
        }
        #pragma unroll
        for (int kk = LOWK; kk < HIK; ++kk) {   // kh = 14, 15
            const v16bf a = load_a(LOWK + kk);
            const v16bf bv0 = load_b_g(ghi + (size_t)(nt0 * HIK + kk) * NFRAG_ELEMS + lane * 8);
            const v16bf bv1 = load_b_g(ghi + (size_t)(nt1 * HIK + kk) * NFRAG_ELEMS + lane * 8);
            acc0 = __builtin_amdgcn_wmma_f32_16x16x32_bf16(false, a, false, bv0,
                                                           (short)0, acc0, false, false);
            acc1 = __builtin_amdgcn_wmma_f32_16x16x32_bf16(false, a, false, bv1,
                                                           (short)0, acc1, false, false);
        }

        // wait for this wave's async noise DMA (hidden behind the whole WMMA phase)
        asm volatile("s_wait_asynccnt 0x0" ::: "memory");

        // -------- Phase 3: fused h update on owned D elements --------
        #pragma unroll
        for (int v = 0; v < 8; ++v) {
            const int row = v + 8 * hi;
            const float i0 = in_stage[row * I_SZ + 0];
            const float i1 = in_stage[row * I_SZ + 1];
            const float i2 = in_stage[row * I_SZ + 2];
            const float i3 = in_stage[row * I_SZ + 3];
            {
                const int col = c0 + m;
                const float inp = i0 * w_im[col] + i1 * w_im[H_SZ + col]
                                + i2 * w_im[2 * H_SZ + col] + i3 * w_im[3 * H_SZ + col];
                const float nz   = noise_stage[row * NSTR + col];
                const float hold = h_lds[row * HSTRIDE + col];
                h_lds[row * HSTRIDE + col] =
                    hold * (1.f - ALPHA_C) + ALPHA_C * (acc0[v] + inp) + NSTD_C * nz;
            }
            {
                const int col = c1 + m;
                const float inp = i0 * w_im[col] + i1 * w_im[H_SZ + col]
                                + i2 * w_im[2 * H_SZ + col] + i3 * w_im[3 * H_SZ + col];
                const float nz   = noise_stage[row * NSTR + col];
                const float hold = h_lds[row * HSTRIDE + col];
                h_lds[row * HSTRIDE + col] =
                    hold * (1.f - ALPHA_C) + ALPHA_C * (acc1[v] + inp) + NSTD_C * nz;
            }
        }
        __syncthreads();
    }
}

extern "C" void kernel_launch(void* const* d_in, const int* in_sizes, int n_in,
                              void* d_out, int out_size, void* d_ws, size_t ws_size,
                              hipStream_t stream) {
    (void)in_sizes; (void)n_in; (void)out_size; (void)ws_size;
    const float* input = (const float*)d_in[0];
    const float* noise = (const float*)d_in[1];
    const float* w_i   = (const float*)d_in[2];
    const float* s_i   = (const float*)d_in[3];
    const float* w_rec = (const float*)d_in[4];
    const float* w_o   = (const float*)d_in[5];
    const float* s_o   = (const float*)d_in[6];
    const float* h0    = (const float*)d_in[7];
    __bf16* wrec_swz   = (__bf16*)d_ws;        // 512 KB scratch

    hipLaunchKernelGGL(wrec_swizzle,
                       dim3((H_SZ * H_SZ + 255) / 256), dim3(256), 0, stream,
                       w_rec, wrec_swz);

    hipLaunchKernelGGL(rnn_persistent,
                       dim3(B_TOT / 16), dim3(512), 0, stream,
                       input, noise, w_i, s_i, w_o, s_o, h0, wrec_swz,
                       (float*)d_out);
}
